// BasicPGCBlock_19662360281496
// MI455X (gfx1250) — matrix operand order repaired and checked by reading, not compiled, not run on hardware
//
#include <hip/hip_runtime.h>

// ---------------------------------------------------------------------------
// BasicPGCBlock on gfx1250:
//   Stage 1: per-pixel Gaussian smooth (sigma = cubic(pmap), 5x5 taps)
//            -> writes sm as NHWC bf16 (hi) + bf16 (lo) split  (fp32-ish precision)
//   Stage 2: 3x3 dilated(2) conv 256->256 + bias + ReLU as implicit GEMM
//            via v_wmma_f32_16x16x32_bf16 with 3-term hi/lo product
// ---------------------------------------------------------------------------

#define HWPIX 9216   // 96*96
#define IMG   96
#define NCH   256

typedef __attribute__((ext_vector_type(16))) __bf16 v16bf;
typedef __attribute__((ext_vector_type(8)))  float  v8f;

union Frag16 {
  v16bf v;
  uint4 u[2];
};

__device__ __forceinline__ unsigned short f2bf(float f) {
  unsigned int u = __float_as_uint(f);
  u += 0x7fffu + ((u >> 16) & 1u);          // round-to-nearest-even
  return (unsigned short)(u >> 16);
}
__device__ __forceinline__ float bf2f(unsigned short h) {
  return __uint_as_float(((unsigned int)h) << 16);
}

// ---------------------------------------------------------------------------
// Pack conv_w [O=256][I=256][3][3] f32 into WMMA A-fragments (hi/lo bf16).
// A-fragment layout (16-bit A 16x32, wave32):
//   lane l (0..31), slot s (0..15):  M = l%16,
//   K = ((l&16)?16:0) + s
// Fragment index: (((tap*8 + kb)*16 + ob)*32 + l)*16 + s  == flat thread id.
// ---------------------------------------------------------------------------
__global__ __launch_bounds__(256) void pack_w(const float* __restrict__ conv_w,
                                              unsigned short* __restrict__ aph,
                                              unsigned short* __restrict__ apl) {
  unsigned tid = blockIdx.x * 256u + threadIdx.x;   // < 589824 = 9*8*16*32*16
  unsigned s  = tid & 15u;
  unsigned l  = (tid >> 4) & 31u;
  unsigned ob = (tid >> 9) & 15u;
  unsigned kb = (tid >> 13) & 7u;
  unsigned t  = tid >> 16;                          // tap 0..8
  unsigned k  = ((l & 16u) ? 16u : 0u) + s;
  unsigned o  = ob * 16u + (l & 15u);
  unsigned i  = kb * 32u + k;
  float w = conv_w[(o * 256u + i) * 9u + t];        // [O][I][3][3], tap = ky*3+kx
  unsigned short hi = f2bf(w);
  unsigned short lo = f2bf(w - bf2f(hi));
  aph[tid] = hi;
  apl[tid] = lo;
}

// ---------------------------------------------------------------------------
// Per-pixel Gaussian smooth.  Block = 8x8 pixel tile of one batch image,
// 256 threads.  Weights (25/pixel) live in LDS; channels processed 32 at a
// time; lane -> channel so the NHWC bf16 hi/lo stores are contiguous.
// ---------------------------------------------------------------------------
__global__ __launch_bounds__(256) void smooth(const float* __restrict__ x,
                                              const float* __restrict__ pmap,
                                              const float* __restrict__ coeffs,
                                              unsigned short* __restrict__ smh,
                                              unsigned short* __restrict__ sml) {
  __shared__ float wts[64 * 25];
  __shared__ float xs[32 * 144];    // 32 channels x 12x12 halo tile

  unsigned bx = blockIdx.x;         // 8 * 12 * 12 = 1152
  unsigned b  = bx / 144u;
  unsigned r  = bx % 144u;
  int h0 = (int)(r / 12u) * 8;
  int w0 = (int)(r % 12u) * 8;
  unsigned t = threadIdx.x;

  if (t < 64u) {
    int py = (int)(t >> 3), px = (int)(t & 7u);
    float p = pmap[b * HWPIX + (unsigned)(h0 + py) * IMG + (unsigned)(w0 + px)];
    float sg = ((coeffs[0] * p + coeffs[1]) * p + coeffs[2]) * p + coeffs[3];
    sg = fmaxf(sg, 0.5f);
    float inv2s2 = 1.0f / (2.0f * sg * sg);
    float wreg[25];
    float sum = 0.f;
#pragma unroll
    for (int j = 0; j < 25; ++j) {
      int u = j / 5 - 2, v = j % 5 - 2;
      float e = __expf(-inv2s2 * (float)(u * u + v * v));
      wreg[j] = e;
      sum += e;
    }
    float inv = 1.0f / sum;
#pragma unroll
    for (int j = 0; j < 25; ++j) wts[t * 25u + j] = wreg[j] * inv;
  }

  const unsigned cc  = t & 31u;       // channel within group (lane -> channel)
  const unsigned pg0 = (t >> 5) * 8u; // 8 pixels per thread

  for (unsigned cg = 0; cg < 8u; ++cg) {
    __syncthreads();
    for (unsigned e = t; e < 4608u; e += 256u) {
      unsigned c    = e / 144u;
      unsigned rr   = (e % 144u) / 12u;
      unsigned colc = e % 12u;
      int gh = h0 - 2 + (int)rr;
      int gw = w0 - 2 + (int)colc;
      float v = 0.f;
      if ((unsigned)gh < 96u && (unsigned)gw < 96u)
        v = x[((b * 256u + cg * 32u + c) * IMG + (unsigned)gh) * IMG + (unsigned)gw];
      xs[e] = v;
    }
    __syncthreads();
#pragma unroll
    for (unsigned q = 0; q < 8u; ++q) {
      unsigned p  = pg0 + q;
      unsigned py = p >> 3, px = p & 7u;
      const float* wp = &wts[p * 25u];
      const float* xp = &xs[cc * 144u + py * 12u + px];
      float acc = 0.f;
#pragma unroll
      for (int j = 0; j < 25; ++j)
        acc = fmaf(wp[j], xp[(j / 5) * 12 + (j % 5)], acc);
      unsigned short hi = f2bf(acc);
      unsigned short lo = f2bf(acc - bf2f(hi));
      unsigned idx = (b * HWPIX + (unsigned)(h0 + (int)py) * IMG +
                      (unsigned)(w0 + (int)px)) * 256u + cg * 32u + cc;
      smh[idx] = hi;
      sml[idx] = lo;
    }
  }
}

// ---------------------------------------------------------------------------
// Implicit-GEMM dilated conv via WMMA.
// Block = 4 waves, 16-pixel row tile; wave w owns output channels [64w,64w+64).
// B fragment (32x16 bf16): lane = pixel (N=l%16), 16 consecutive channels
// (K = (l>=16?16:0)+s) -> one contiguous 32B load per lane from NHWC sm.
// Per (tap,kblock): 1 B-pair load, reused by 4 A-tiles; 3 WMMAs per tile
// for the hi/lo split:  Ah*Bh + Ah*Bl + Al*Bh.
// ---------------------------------------------------------------------------
__global__ __launch_bounds__(128) void conv_wmma(
    const unsigned short* __restrict__ smh, const unsigned short* __restrict__ sml,
    const unsigned short* __restrict__ aph, const unsigned short* __restrict__ apl,
    const float* __restrict__ bias, float* __restrict__ out) {
  unsigned bx = blockIdx.x;          // 8 * 96 * 6 = 4608
  unsigned b  = bx / 576u;
  unsigned r  = bx % 576u;
  int h  = (int)(r / 6u);
  int w0 = (int)(r % 6u) * 16;
  unsigned tid    = threadIdx.x;
  unsigned wave   = tid >> 5;        // 0..3
  unsigned lane   = tid & 31u;
  unsigned n      = lane & 15u;      // pixel within tile
  unsigned hiHalf = lane >> 4;       // 0/1

  v8f acc[4] = {};

  for (int tap = 0; tap < 9; ++tap) {
    int dy = (tap / 3 - 1) * 2;
    int dx = (tap % 3 - 1) * 2;
    int hh = h + dy;
    if ((unsigned)hh >= 96u) continue;            // wave-uniform
    int ww = w0 + (int)n + dx;
    bool ok = (unsigned)ww < 96u;                 // per-lane column check
    unsigned pixBase = ok ? ((b * HWPIX + (unsigned)hh * IMG + (unsigned)ww) * 256u) : 0u;

    for (unsigned kb = 0; kb < 8u; ++kb) {
      unsigned coff = kb * 32u + hiHalf * 16u;
      Frag16 bh, bl;
      bh.u[0] = bh.u[1] = bl.u[0] = bl.u[1] = make_uint4(0u, 0u, 0u, 0u);
      if (ok) {
        const uint4* ph = (const uint4*)(smh + pixBase + coff);
        bh.u[0] = ph[0];
        bh.u[1] = ph[1];
        const uint4* pl = (const uint4*)(sml + pixBase + coff);
        bl.u[0] = pl[0];
        bl.u[1] = pl[1];
      }
      unsigned fbase = ((unsigned)tap * 8u + kb) * 16u;
#pragma unroll
      for (unsigned ot = 0; ot < 4u; ++ot) {
        unsigned otile = wave * 4u + ot;
        unsigned foff  = ((fbase + otile) * 32u + lane) * 16u;
        Frag16 ah, al;
        const uint4* pa = (const uint4*)(aph + foff);
        ah.u[0] = pa[0];
        ah.u[1] = pa[1];
        const uint4* pb = (const uint4*)(apl + foff);
        al.u[0] = pb[0];
        al.u[1] = pb[1];
        acc[ot] = __builtin_amdgcn_wmma_f32_16x16x32_bf16(
            false, ah.v, false, bh.v, (short)0, acc[ot], false, false);
        acc[ot] = __builtin_amdgcn_wmma_f32_16x16x32_bf16(
            false, ah.v, false, bl.v, (short)0, acc[ot], false, false);
        acc[ot] = __builtin_amdgcn_wmma_f32_16x16x32_bf16(
            false, al.v, false, bh.v, (short)0, acc[ot], false, false);
      }
    }
  }

  // Epilogue: C/D layout -> row m = v + 8*hiHalf, col n = lane%16
  unsigned pix = (unsigned)h * IMG + (unsigned)w0 + n;
#pragma unroll
  for (unsigned ot = 0; ot < 4u; ++ot) {
    unsigned ob = (wave * 4u + ot) * 16u;
#pragma unroll
    for (unsigned v = 0; v < 8u; ++v) {
      unsigned o = ob + v + hiHalf * 8u;
      float val = acc[ot][v] + bias[o];
      out[(b * 256u + o) * HWPIX + pix] = fmaxf(val, 0.f);
    }
  }
}

// ---------------------------------------------------------------------------
// Launch.  Workspace layout (bytes):
//   [0,            37748736)   smh  : 8*96*96*256 bf16  (NHWC hi)
//   [37748736,     75497472)   sml  : same            (NHWC lo)
//   [75497472,     76677120)   aph  : 9*8*16*512 bf16  (packed A hi)
//   [76677120,     77856768)   apl  : packed A lo
// ---------------------------------------------------------------------------
extern "C" void kernel_launch(void* const* d_in, const int* in_sizes, int n_in,
                              void* d_out, int out_size, void* d_ws, size_t ws_size,
                              hipStream_t stream) {
  const float* x      = (const float*)d_in[0];
  const float* pmap   = (const float*)d_in[1];
  const float* coeffs = (const float*)d_in[2];
  const float* conv_w = (const float*)d_in[3];
  const float* conv_b = (const float*)d_in[4];
  float* out = (float*)d_out;

  char* ws = (char*)d_ws;
  unsigned short* smh = (unsigned short*)ws;
  unsigned short* sml = smh + 18874368u;
  unsigned short* aph = (unsigned short*)(ws + 75497472u);
  unsigned short* apl = aph + 589824u;

  pack_w<<<2304, 256, 0, stream>>>(conv_w, aph, apl);
  smooth<<<1152, 256, 0, stream>>>(x, pmap, coeffs, smh, sml);
  conv_wmma<<<4608, 128, 0, stream>>>(smh, sml, aph, apl, conv_b, out);

  (void)in_sizes; (void)n_in; (void)out_size; (void)ws_size;
}
